// GAT_41506563949060
// MI455X (gfx1250) — matrix operand order repaired and checked by reading, not compiled, not run on hardware
//
#include <hip/hip_runtime.h>
#include <hip/hip_bf16.h>
#include <math.h>

typedef __attribute__((ext_vector_type(16))) __bf16 v16bf;
typedef __attribute__((ext_vector_type(8)))  float  v8f;

#define FIN   128
#define HID   64
#define HEADS 4
#define CH    16
#define FOUT  40
#define NEG_SLOPE 0.2f

// -------- float atomic max via sign-aware integer atomics (IEEE order trick) ----
__device__ __forceinline__ void atomicMaxF(float* addr, float v) {
  if (v >= 0.0f) atomicMax((int*)addr, __float_as_int(v));
  else           atomicMin((unsigned int*)addr, __float_as_uint(v));
}

// -------- pack W [K, Ncols] fp32 -> bf16 B-fragment layout Wp[n][K/16][16].
// Lane with column n and k-block kk reads 16 contiguous bf16 (32B) per WMMA step.
// Columns n >= Ncols (padding up to Np, multiple of 16) are zero-filled.
__global__ void pack_W_bf16(const float* __restrict__ W, __bf16* __restrict__ Wp,
                            int K, int Ncols, int Np) {
  int idx = blockIdx.x * blockDim.x + threadIdx.x;   // idx = n*K + k
  if (idx >= Np * K) return;
  int n = idx / K;
  int k = idx - n * K;
  float v = (n < Ncols) ? W[(size_t)k * Ncols + n] : 0.0f;
  // destination ((n*(K/16) + k/16)*16 + k%16) == n*K + k (same flat index)
  Wp[idx] = (__bf16)v;
}

// -------- C[M,Ncols] = A[M,K](fp32) * W(packed bf16), fp32 accumulate.
// One wave per 16-row tile; the wave loops over all NT column tiles, keeping the
// A fragment in registers (A read exactly once). blockDim = 128 (4 waves).
// REQUIRES: M % 16 == 0, K % 32 == 0, Np = NT*16 >= Ncols.
template<int NT>
__global__ void wmma_gemm_packed(const float* __restrict__ A,
                                 const __bf16* __restrict__ Wp,
                                 float* __restrict__ C,
                                 int M, int K, int Ncols) {
  const int lane = threadIdx.x & 31;
  const int wave = threadIdx.x >> 5;
  const int half = lane >> 4;          // 0: lanes 0-15, 1: lanes 16-31
  const int l16  = lane & 15;
  const int row0 = (blockIdx.x * 4 + wave) << 4;
  if (row0 >= M) return;               // uniform per wave: no WMMA under partial EXEC
  const int m    = row0 + l16;         // A row held by this lane (m < M since M%16==0)
  const int Ksub = K >> 4;

  v8f acc[NT];
  #pragma unroll
  for (int t = 0; t < NT; ++t) acc[t] = v8f{};

  const float* arow = A + (size_t)m * K;
  for (int k0 = 0; k0 < K; k0 += 32) {
    // A fragment: K = k0 + half*16 + {0..15} (contiguous, matches packed B)
    const float* ab = arow + k0 + (half << 4);
    __builtin_prefetch(ab + 32, 0, 1);              // next k-block (global_prefetch_b8)
    float4 f0 = *(const float4*)(ab + 0);
    float4 f1 = *(const float4*)(ab + 4);
    float4 f2 = *(const float4*)(ab + 8);
    float4 f3 = *(const float4*)(ab + 12);
    v16bf a;
    a[0]=(__bf16)f0.x;  a[1]=(__bf16)f0.y;  a[2]=(__bf16)f0.z;  a[3]=(__bf16)f0.w;
    a[4]=(__bf16)f1.x;  a[5]=(__bf16)f1.y;  a[6]=(__bf16)f1.z;  a[7]=(__bf16)f1.w;
    a[8]=(__bf16)f2.x;  a[9]=(__bf16)f2.y;  a[10]=(__bf16)f2.z; a[11]=(__bf16)f2.w;
    a[12]=(__bf16)f3.x; a[13]=(__bf16)f3.y; a[14]=(__bf16)f3.z; a[15]=(__bf16)f3.w;

    const int kk = (k0 >> 4) + half;   // this lane's K/16 block (B: K = kk*16 + e)
    #pragma unroll
    for (int t = 0; t < NT; ++t) {
      const int n = (t << 4) + l16;    // B column held by this lane
      v16bf b = *(const v16bf*)(Wp + (n * Ksub + kk) * 16);
      acc[t] = __builtin_amdgcn_wmma_f32_16x16x32_bf16(false, a, false, b,
                                                       (short)0, acc[t], false, false);
    }
  }

  // D row = vgpr + half*8 (ISA 7.12.2). 32-bit offsets: M*Ncols <= 6.4M here.
  const int rbase = (row0 + (half << 3)) * Ncols;
  #pragma unroll
  for (int t = 0; t < NT; ++t) {
    const int n = (t << 4) + l16;
    if (n < Ncols) {
      const int cb = rbase + n;
      #pragma unroll
      for (int v = 0; v < 8; ++v)
        C[cb + v * Ncols] = acc[t][v];
    }
  }
}

// -------- per-node attention logits: e_src/e_dst[n,h] = <h[n,h,:], a_src/a_dst[h,:]>
template<int H, int C>
__global__ void attn_logits(const float* __restrict__ h,
                            const float* __restrict__ a_src,
                            const float* __restrict__ a_dst,
                            float* __restrict__ e_src,
                            float* __restrict__ e_dst,
                            int N) {
  int idx = blockIdx.x * blockDim.x + threadIdx.x;   // idx = n*H + hh
  if (idx >= N * H) return;
  int hh = idx & (H - 1);                            // H is a power of two (4 or 1)
  const float* hp = h + (size_t)idx * C;
  float s = 0.f, d = 0.f;
  #pragma unroll
  for (int c = 0; c < C; ++c) {
    float v = hp[c];
    s = fmaf(v, a_src[hh * C + c], s);
    d = fmaf(v, a_dst[hh * C + c], d);
  }
  e_src[idx] = s;
  e_dst[idx] = d;
}

// -------- init segment buffers: m=-inf, denom=0, acc=0
__global__ void init_layer(float* __restrict__ m, float* __restrict__ denom,
                           float* __restrict__ acc, int NH, int NF) {
  int idx = blockIdx.x * blockDim.x + threadIdx.x;
  if (idx < NH) { m[idx] = -INFINITY; denom[idx] = 0.0f; }
  if (idx < NF) acc[idx] = 0.0f;
}

__device__ __forceinline__ void edge_ends(const long long* ei, int E, int eid,
                                          int* s, int* d) {
  *s = (eid < E) ? (int)ei[eid]     : (eid - E);     // self-loops appended
  *d = (eid < E) ? (int)ei[E + eid] : (eid - E);
}

// -------- pass 1: segment max over dst
template<int H>
__global__ void edge_max(const long long* __restrict__ ei, int E, int N,
                         const float* __restrict__ e_src, const float* __restrict__ e_dst,
                         float* __restrict__ m) {
  int total = (E + N) * H;
  for (int idx = blockIdx.x * blockDim.x + threadIdx.x; idx < total;
       idx += gridDim.x * blockDim.x) {
    int eid = idx / H, hh = idx & (H - 1);
    int s, d; edge_ends(ei, E, eid, &s, &d);
    float ev = e_src[s * H + hh] + e_dst[d * H + hh];
    ev = ev > 0.0f ? ev : NEG_SLOPE * ev;            // leaky_relu(0.2)
    atomicMaxF(&m[d * H + hh], ev);
  }
}

// -------- pass 2: segment sum of exp(e - m[dst])
template<int H>
__global__ void edge_sum(const long long* __restrict__ ei, int E, int N,
                         const float* __restrict__ e_src, const float* __restrict__ e_dst,
                         const float* __restrict__ m, float* __restrict__ denom) {
  int total = (E + N) * H;
  for (int idx = blockIdx.x * blockDim.x + threadIdx.x; idx < total;
       idx += gridDim.x * blockDim.x) {
    int eid = idx / H, hh = idx & (H - 1);
    int s, d; edge_ends(ei, E, eid, &s, &d);
    float ev = e_src[s * H + hh] + e_dst[d * H + hh];
    ev = ev > 0.0f ? ev : NEG_SLOPE * ev;
    atomicAdd(&denom[d * H + hh], __expf(ev - m[d * H + hh]));
  }
}

// -------- pass 3: gather h[src], scale by alpha, scatter-add into acc[dst]
template<int H, int C>
__global__ void edge_scatter(const long long* __restrict__ ei, int E, int N,
                             const float* __restrict__ h,
                             const float* __restrict__ e_src, const float* __restrict__ e_dst,
                             const float* __restrict__ m, const float* __restrict__ denom,
                             float* __restrict__ acc) {
  int total = (E + N) * H;
  for (int idx = blockIdx.x * blockDim.x + threadIdx.x; idx < total;
       idx += gridDim.x * blockDim.x) {
    int eid = idx / H, hh = idx & (H - 1);
    int s, d; edge_ends(ei, E, eid, &s, &d);
    float ev = e_src[s * H + hh] + e_dst[d * H + hh];
    ev = ev > 0.0f ? ev : NEG_SLOPE * ev;
    int dh = d * H + hh;
    float alpha = __expf(ev - m[dh]) / (denom[dh] + 1e-16f);
    const float4* hs = (const float4*)(h + ((size_t)s * H + hh) * C);
    float* ad = acc + ((size_t)d * H + hh) * C;
    #pragma unroll
    for (int c = 0; c < C; c += 4) {
      float4 v = hs[c >> 2];
      atomicAdd(ad + c + 0, v.x * alpha);
      atomicAdd(ad + c + 1, v.y * alpha);
      atomicAdd(ad + c + 2, v.z * alpha);
      atomicAdd(ad + c + 3, v.w * alpha);
    }
  }
}

// -------- layers 1-3 epilogue: += bias (F = HID compile-time)
template<int F>
__global__ void add_bias(float* __restrict__ acc, const float* __restrict__ b, int N) {
  int idx = blockIdx.x * blockDim.x + threadIdx.x;
  if (idx >= N * F) return;
  acc[idx] += b[idx & (F - 1)];                      // F = 64, power of two
}

// -------- layer 4 epilogue: out = log_softmax(acc + b), F_out = 40 fixed
__global__ void logsoftmax_bias(const float* __restrict__ acc, const float* __restrict__ b,
                                float* __restrict__ out, int N) {
  int n = blockIdx.x * blockDim.x + threadIdx.x;
  if (n >= N) return;
  float v[FOUT];
  float mx = -INFINITY;
  #pragma unroll
  for (int c = 0; c < FOUT; ++c) {
    v[c] = acc[(size_t)n * FOUT + c] + b[c];
    mx = fmaxf(mx, v[c]);
  }
  float s = 0.f;
  #pragma unroll
  for (int c = 0; c < FOUT; ++c) s += __expf(v[c] - mx);
  float lse = mx + __logf(s);
  #pragma unroll
  for (int c = 0; c < FOUT; ++c) out[(size_t)n * FOUT + c] = v[c] - lse;
}

extern "C" void kernel_launch(void* const* d_in, const int* in_sizes, int n_in,
                              void* d_out, int out_size, void* d_ws, size_t ws_size,
                              hipStream_t stream) {
  const float*     x  = (const float*)d_in[0];
  const long long* ei = (const long long*)d_in[1];
  const float* Wm[4] = {(const float*)d_in[2],  (const float*)d_in[6],
                        (const float*)d_in[10], (const float*)d_in[14]};
  const float* As[4] = {(const float*)d_in[3],  (const float*)d_in[7],
                        (const float*)d_in[11], (const float*)d_in[15]};
  const float* Ad[4] = {(const float*)d_in[4],  (const float*)d_in[8],
                        (const float*)d_in[12], (const float*)d_in[16]};
  const float* Bv[4] = {(const float*)d_in[5],  (const float*)d_in[9],
                        (const float*)d_in[13], (const float*)d_in[17]};

  const int N = in_sizes[0] / FIN;    // 100000 (multiple of 16)
  const int E = in_sizes[1] / 2;      // 1600000

  // ---- carve workspace ----
  char* ws = (char*)d_ws;
  size_t off = 0;
  auto carve = [&](size_t bytes) -> void* {
    void* p = ws + off;
    off = (off + bytes + 255) & ~(size_t)255;
    return p;
  };
  float*  hA = (float*)carve((size_t)N * HID * sizeof(float));   // projected features
  float*  hB = (float*)carve((size_t)N * HID * sizeof(float));   // accumulator / next input
  float*  eS = (float*)carve((size_t)N * HEADS * sizeof(float));
  float*  eD = (float*)carve((size_t)N * HEADS * sizeof(float));
  float*  mM = (float*)carve((size_t)N * HEADS * sizeof(float));
  float*  dn = (float*)carve((size_t)N * HEADS * sizeof(float));
  __bf16* Wp = (__bf16*)carve((size_t)FIN * HID * sizeof(__bf16)); // packed W (max 16KB)

  const int rowTiles = (N + 15) / 16;
  const int gemmGrid = (rowTiles + 3) / 4;           // 4 waves (row tiles) per block

  auto layer = [&](const float* X, int K, int li, bool last) {
    const int H  = last ? 1 : HEADS;
    const int C  = last ? FOUT : CH;
    const int F  = H * C;
    const int Np = (F + 15) & ~15;
    const int NH = N * H;
    const int NF = N * F;
    const int tot = (E + N) * H;
    const int blk = (tot + 255) / 256;

    pack_W_bf16<<<(Np * K + 255) / 256, 256, 0, stream>>>(Wm[li], Wp, K, F, Np);

    if (!last) {
      wmma_gemm_packed<4><<<gemmGrid, 128, 0, stream>>>(X, Wp, hA, N, K, F);
      attn_logits<HEADS, CH><<<(NH + 255) / 256, 256, 0, stream>>>(hA, As[li], Ad[li],
                                                                   eS, eD, N);
      init_layer<<<(NF + 255) / 256, 256, 0, stream>>>(mM, dn, hB, NH, NF);
      edge_max<HEADS>    <<<blk, 256, 0, stream>>>(ei, E, N, eS, eD, mM);
      edge_sum<HEADS>    <<<blk, 256, 0, stream>>>(ei, E, N, eS, eD, mM, dn);
      edge_scatter<HEADS, CH><<<blk, 256, 0, stream>>>(ei, E, N, hA, eS, eD, mM, dn, hB);
      add_bias<HID><<<(NF + 255) / 256, 256, 0, stream>>>(hB, Bv[li], N);
    } else {
      wmma_gemm_packed<3><<<gemmGrid, 128, 0, stream>>>(X, Wp, hA, N, K, F);
      attn_logits<1, FOUT><<<(NH + 255) / 256, 256, 0, stream>>>(hA, As[li], Ad[li],
                                                                 eS, eD, N);
      init_layer<<<(NF + 255) / 256, 256, 0, stream>>>(mM, dn, hB, NH, NF);
      edge_max<1>    <<<blk, 256, 0, stream>>>(ei, E, N, eS, eD, mM);
      edge_sum<1>    <<<blk, 256, 0, stream>>>(ei, E, N, eS, eD, mM, dn);
      edge_scatter<1, FOUT><<<blk, 256, 0, stream>>>(ei, E, N, hA, eS, eD, mM, dn, hB);
      logsoftmax_bias<<<(N + 255) / 256, 256, 0, stream>>>(hB, Bv[li], (float*)d_out, N);
    }
  };

  layer(x,  FIN, 0, false);   // 128 -> 64, 4 heads, concat
  layer(hB, HID, 1, false);   // 64 -> 64, 4 heads, concat
  layer(hB, HID, 2, false);   // 64 -> 64, 4 heads, concat
  layer(hB, HID, 3, true);    // 64 -> 40, single head, log_softmax
}
